// _UnidirectionalMambaBlock_4544075399317
// MI455X (gfx1250) — matrix operand order repaired and checked by reading, not compiled, not run on hardware
//
#include <hip/hip_runtime.h>
#include <hip/hip_bf16.h>
#include <stdint.h>

typedef __bf16 bf16;
typedef __attribute__((ext_vector_type(16))) __bf16 v16bf;
typedef __attribute__((ext_vector_type(8)))  float  v8f;

// Problem constants
#define BB      8
#define LL      1024
#define DD      512
#define DSTATE  16
#define DCONV   4
#define DTRANK  32
#define DIN     1024
#define MM      (BB*LL)             // 8192 rows
#define NXZ     (2*DIN)             // 2048
#define NDBL    (DTRANK + 2*DSTATE) // 64

// ---------------- gfx1250 async global->LDS copy (guarded, safe fallback) ------------
#if defined(__HIP_DEVICE_COMPILE__) && __has_builtin(__builtin_amdgcn_global_load_async_to_lds_b128)
#define USE_ASYNC_LDS 1
#endif

#ifdef USE_ASYNC_LDS
// Matches the builtin's parameter type revealed by the compiler diagnostic:
//   'int __attribute__((__vector_size__(4 * sizeof(int)))) __device__ *'
typedef int async_v4i __attribute__((__vector_size__(4 * sizeof(int))));
typedef __attribute__((address_space(1))) async_v4i* g_v4i_ptr;
typedef __attribute__((address_space(3))) async_v4i* l_v4i_ptr;
#endif

__device__ __forceinline__ void copy16_g2l(const bf16* g, bf16* l) {
#ifdef USE_ASYNC_LDS
    __builtin_amdgcn_global_load_async_to_lds_b128(
        (g_v4i_ptr)(uintptr_t)g,
        (l_v4i_ptr)(uintptr_t)l,
        0, 0);
#else
    *(uint4*)l = *(const uint4*)g;
#endif
}

__device__ __forceinline__ void async_copies_wait() {
#ifdef USE_ASYNC_LDS
#if __has_builtin(__builtin_amdgcn_s_wait_asynccnt)
    __builtin_amdgcn_s_wait_asynccnt(0);
#else
    asm volatile("s_wait_asynccnt 0x0" ::: "memory");
#endif
#endif
}

__device__ inline float wave_sum32(float v) {
    #pragma unroll
    for (int off = 16; off > 0; off >>= 1) v += __shfl_xor(v, off, 32);
    return v;
}

// ---------------- weight conversion: fp32 (KxN) -> bf16 transposed (NxK) ----------------
__global__ void cvt_transpose_bf16_kernel(const float* __restrict__ in, bf16* __restrict__ out,
                                          int K, int N) {
    int idx = blockIdx.x * blockDim.x + threadIdx.x;   // index into out (NxK)
    if (idx >= K * N) return;
    int n = idx / K, k = idx % K;
    out[idx] = (bf16)in[(size_t)k * N + n];
}

__global__ void neg_exp_kernel(const float* __restrict__ in, float* __restrict__ out, int n) {
    int i = blockIdx.x * blockDim.x + threadIdx.x;
    if (i < n) out[i] = -__expf(in[i]);
}

// ---------------- LayerNorm 1 (fp32 -> bf16), one wave per row of 512 ----------------
__global__ __launch_bounds__(256) void ln1_kernel(const float* __restrict__ x,
                                                  const float* __restrict__ w,
                                                  const float* __restrict__ b,
                                                  bf16* __restrict__ out) {
    const int lane = threadIdx.x & 31;
    const int wave = threadIdx.x >> 5;
    const int row  = blockIdx.x * 8 + wave;
    const float* xr = x + (size_t)row * DD;
    float v[16], s = 0.f, sq = 0.f;
    #pragma unroll
    for (int i = 0; i < 16; ++i) {
        v[i] = xr[lane + i * 32];
        s += v[i]; sq += v[i] * v[i];
    }
    s  = wave_sum32(s);
    sq = wave_sum32(sq);
    const float mean = s * (1.f / DD);
    const float var  = sq * (1.f / DD) - mean * mean;
    const float rstd = rsqrtf(var + 1e-5f);
    #pragma unroll
    for (int i = 0; i < 16; ++i) {
        int c = lane + i * 32;
        out[(size_t)row * DD + c] = (bf16)((v[i] - mean) * rstd * w[c] + b[c]);
    }
}

// ---------------- Residual + LayerNorm 2 (fp32 out) ----------------
__global__ __launch_bounds__(256) void ln2_kernel(const float* __restrict__ x,
                                                  const float* __restrict__ h,
                                                  const float* __restrict__ w,
                                                  const float* __restrict__ b,
                                                  float* __restrict__ out) {
    const int lane = threadIdx.x & 31;
    const int wave = threadIdx.x >> 5;
    const int row  = blockIdx.x * 8 + wave;
    const size_t base = (size_t)row * DD;
    float v[16], s = 0.f, sq = 0.f;
    #pragma unroll
    for (int i = 0; i < 16; ++i) {
        int c = lane + i * 32;
        v[i] = x[base + c] + h[base + c];
        s += v[i]; sq += v[i] * v[i];
    }
    s  = wave_sum32(s);
    sq = wave_sum32(sq);
    const float mean = s * (1.f / DD);
    const float var  = sq * (1.f / DD) - mean * mean;
    const float rstd = rsqrtf(var + 1e-5f);
    #pragma unroll
    for (int i = 0; i < 16; ++i) {
        int c = lane + i * 32;
        out[base + c] = (v[i] - mean) * rstd * w[c] + b[c];
    }
}

// ---------------- WMMA bf16 GEMM, double-buffered async pipeline ----------------
// C (MxN fp32) = A (MxK bf16 row-major) * B, where B is provided TRANSPOSED: Bt (NxK bf16 row-major).
// BM = 128, BK = 32, 8 waves / block. Tiles staged via async 16B global->LDS copies,
// tile k+1 in flight while tile k feeds the WMMA pipes; one barrier per K-step.
template<int BM, int BN>
__global__ __launch_bounds__(256) void gemm_bf16_wmma(const bf16* __restrict__ A,
                                                      const bf16* __restrict__ Bt,
                                                      float* __restrict__ C,
                                                      int M, int N, int K) {
    constexpr int BK  = 32;
    constexpr int PAD = 8;
    constexpr int WAVES_N = (BN >= 128) ? 2 : 1;
    constexpr int WAVES_M = 8 / WAVES_N;
    constexpr int TM = BM / WAVES_M;      // 32 or 16
    constexpr int TN = BN / WAVES_N;      // 64
    constexpr int FM = TM / 16;           // 2 or 1
    constexpr int FN = TN / 16;           // 4
    constexpr int AVEC = (BM * BK) / (256 * 8);  // 2
    constexpr int BVEC = (BN * BK) / (256 * 8);  // 2 or 1

    __shared__ bf16 As[2][BM][BK + PAD];
    __shared__ bf16 Bs[2][BN][BK + PAD];

    const int tid  = threadIdx.x;
    const int lane = tid & 31;
    const int wave = tid >> 5;
    const int wm   = wave / WAVES_N;
    const int wn   = wave % WAVES_N;
    const int m0   = blockIdx.y * BM;
    const int n0   = blockIdx.x * BN;

    auto stage = [&](int k0, int buf) {
        #pragma unroll
        for (int v = 0; v < AVEC; ++v) {
            int e   = (tid + v * 256) * 8;
            int row = e / BK;
            int kk  = e % BK;
            copy16_g2l(&A[(size_t)(m0 + row) * K + k0 + kk], &As[buf][row][kk]);
        }
        #pragma unroll
        for (int v = 0; v < BVEC; ++v) {
            int e  = (tid + v * 256) * 8;
            int n  = e / BK;
            int kk = e % BK;
            copy16_g2l(&Bt[(size_t)(n0 + n) * K + k0 + kk], &Bs[buf][n][kk]);
        }
    };

    v8f acc[FM][FN];
    #pragma unroll
    for (int i = 0; i < FM; ++i)
        #pragma unroll
        for (int j = 0; j < FN; ++j)
            acc[i][j] = (v8f)0.0f;

    const int KT = K / BK;
    stage(0, 0);

    const int half = (lane < 16) ? 0 : 1;
    const int lm   = lane & 15;
    const int klo  = half * 8;    // A-fragment K select (ISA 16-bit A layout)
    const int kb   = half * 16;   // unused; B fragment also uses klo

    for (int kt = 0; kt < KT; ++kt) {
        async_copies_wait();      // own async copies for tile kt retired
        __syncthreads();          // all waves' tile kt visible; tile kt-1 readers done
        if (kt + 1 < KT) stage((kt + 1) * BK, (kt + 1) & 1);
        const int buf = kt & 1;

        v16bf afr[FM], bfr[FN];
        #pragma unroll
        for (int i = 0; i < FM; ++i) {
            const bf16* p = &As[buf][wm * TM + i * 16 + lm][0];
            #pragma unroll
            for (int j = 0; j < 8; ++j) {
                afr[i][j]     = p[klo + j];
                afr[i][8 + j] = p[16 + klo + j];
            }
        }
        #pragma unroll
        for (int i = 0; i < FN; ++i) {
            const bf16* p = &Bs[buf][wn * TN + i * 16 + lm][0];
            #pragma unroll
            for (int j = 0; j < 8; ++j) {
                bfr[i][j]     = p[klo + j];
                bfr[i][8 + j] = p[16 + klo + j];
            }
        }
        #pragma unroll
        for (int i = 0; i < FM; ++i)
            #pragma unroll
            for (int j = 0; j < FN; ++j)
                acc[i][j] = __builtin_amdgcn_wmma_f32_16x16x32_bf16(
                    false, afr[i], false, bfr[j], (short)0, acc[i][j], false, false);
    }

    // Store: VGPR r of v8f holds (M=r, N=lane) for lanes 0-15, (M=8+r, N=lane-16) for 16-31
    #pragma unroll
    for (int i = 0; i < FM; ++i)
        #pragma unroll
        for (int j = 0; j < FN; ++j) {
            const int col   = n0 + wn * TN + j * 16 + (lane & 15);
            const int rbase = m0 + wm * TM + i * 16 + ((lane < 16) ? 0 : 8);
            float* cp = &C[(size_t)rbase * N + col];
            #pragma unroll
            for (int r = 0; r < 8; ++r)
                cp[(size_t)r * N] = acc[i][j][r];
        }
}

// ---------------- causal depthwise conv (DCONV=4) + bias + SiLU ----------------
__global__ void conv_silu_kernel(const float* __restrict__ xz,
                                 const float* __restrict__ cw,
                                 const float* __restrict__ cb,
                                 float* __restrict__ xp,
                                 bf16* __restrict__ xpb) {
    int idx = blockIdx.x * blockDim.x + threadIdx.x;  // over MM*DIN
    if (idx >= MM * DIN) return;
    int d  = idx % DIN;
    int ml = idx / DIN;       // b*L + l
    int l  = ml % LL;
    float s = cb[d];
    #pragma unroll
    for (int j = 0; j < DCONV; ++j) {
        int t = l - (DCONV - 1) + j;
        if (t >= 0)
            s += xz[(size_t)(ml - (DCONV - 1) + j) * NXZ + d] * cw[d * DCONV + j];
    }
    float v = s / (1.f + __expf(-s));   // SiLU
    xp[idx]  = v;
    xpb[idx] = (bf16)v;
}

// ---------------- extract dt_r (first 32 cols of dbl) to bf16 ----------------
__global__ void extract_dtr_kernel(const float* __restrict__ dbl, bf16* __restrict__ dtr) {
    int idx = blockIdx.x * blockDim.x + threadIdx.x;
    if (idx >= MM * DTRANK) return;
    int r = idx / DTRANK, c = idx % DTRANK;
    dtr[idx] = (bf16)dbl[(size_t)r * NDBL + c];
}

// ---------------- selective scan, fused softplus + D-skip + z gating, bf16 out ----------------
__global__ __launch_bounds__(256) void scan_kernel(const float* __restrict__ dtp,
                                                   const float* __restrict__ xp,
                                                   const float* __restrict__ dbl,
                                                   const float* __restrict__ xz,
                                                   const float* __restrict__ Aneg,
                                                   const float* __restrict__ Dskip,
                                                   const float* __restrict__ bdt,
                                                   bf16* __restrict__ yb) {
    const int d = blockIdx.x * 256 + threadIdx.x;
    const int b = blockIdx.y;
    __shared__ float sBC[2 * DSTATE];

    float Areg[DSTATE];
    #pragma unroll
    for (int n = 0; n < DSTATE; ++n) Areg[n] = Aneg[d * DSTATE + n];
    const float dsk = Dskip[d];
    const float bd  = bdt[d];
    float h[DSTATE];
    #pragma unroll
    for (int n = 0; n < DSTATE; ++n) h[n] = 0.f;

    for (int l = 0; l < LL; ++l) {
        const size_t ml = (size_t)b * LL + l;
        if (threadIdx.x < 2 * DSTATE)
            sBC[threadIdx.x] = dbl[ml * NDBL + DTRANK + threadIdx.x];
        __syncthreads();

        if (l + 8 < LL) {   // hide global latency of the serial loop
            __builtin_prefetch(&dtp[(ml + 8) * DIN + d], 0, 1);
            __builtin_prefetch(&xp[(ml + 8) * DIN + d], 0, 1);
        }

        float dpre = dtp[ml * DIN + d] + bd;
        float dt   = (dpre > 20.f) ? dpre : log1pf(__expf(dpre));
        float xv   = xp[ml * DIN + d];
        float dtx  = dt * xv;
        float acc  = 0.f;
        #pragma unroll
        for (int n = 0; n < DSTATE; ++n) {
            h[n] = __expf(dt * Areg[n]) * h[n] + dtx * sBC[n];
            acc += h[n] * sBC[DSTATE + n];
        }
        float zv = xz[ml * NXZ + DIN + d];
        float yv = (acc + xv * dsk) * (zv / (1.f + __expf(-zv)));
        yb[ml * DIN + d] = (bf16)yv;
        __syncthreads();
    }
}

// ===================================================================================
extern "C" void kernel_launch(void* const* d_in, const int* in_sizes, int n_in,
                              void* d_out, int out_size, void* d_ws, size_t ws_size,
                              hipStream_t stream) {
    const float* x      = (const float*)d_in[0];
    const float* ln1_w  = (const float*)d_in[1];
    const float* ln1_b  = (const float*)d_in[2];
    const float* W_in   = (const float*)d_in[3];
    const float* conv_w = (const float*)d_in[4];
    const float* conv_b = (const float*)d_in[5];
    const float* W_x    = (const float*)d_in[6];
    const float* W_dt   = (const float*)d_in[7];
    const float* b_dt   = (const float*)d_in[8];
    const float* A_log  = (const float*)d_in[9];
    const float* D_skip = (const float*)d_in[10];
    const float* W_out  = (const float*)d_in[11];
    const float* ln2_w  = (const float*)d_in[12];
    const float* ln2_b  = (const float*)d_in[13];
    float* out = (float*)d_out;

    char* ws = (char*)d_ws;
    size_t off = 0;
    auto alloc = [&](size_t bytes) {
        size_t o = off;
        off = (off + bytes + 255) & ~(size_t)255;
        return o;
    };
    bf16*  wWin  = (bf16*)(ws + alloc((size_t)DD * NXZ * 2));    // transposed: NXZ x DD
    bf16*  wWx   = (bf16*)(ws + alloc((size_t)DIN * NDBL * 2));  // transposed: NDBL x DIN
    bf16*  wWdt  = (bf16*)(ws + alloc((size_t)DTRANK * DIN * 2));// transposed: DIN x DTRANK
    bf16*  wWout = (bf16*)(ws + alloc((size_t)DIN * DD * 2));    // transposed: DD x DIN
    float* wA    = (float*)(ws + alloc((size_t)DIN * DSTATE * 4));
    bf16*  xln   = (bf16*)(ws + alloc((size_t)MM * DD * 2));
    float* xzbuf = (float*)(ws + alloc((size_t)MM * NXZ * 4));
    float* xpbuf = (float*)(ws + alloc((size_t)MM * DIN * 4));
    bf16*  xpb   = (bf16*)(ws + alloc((size_t)MM * DIN * 2));
    float* dblb  = (float*)(ws + alloc((size_t)MM * NDBL * 4));
    bf16*  dtr   = (bf16*)(ws + alloc((size_t)MM * DTRANK * 2));
    float* dtp   = (float*)(ws + alloc((size_t)MM * DIN * 4));
    bf16*  ybuf  = (bf16*)(ws + alloc((size_t)MM * DIN * 2));
    float* hbuf  = (float*)(ws + alloc((size_t)MM * DD * 4));
    (void)ws_size; (void)in_sizes; (void)n_in; (void)out_size;

    // 1) weight conversions (fp32 KxN -> bf16 NxK)
    {
        int n;
        n = DD * NXZ;      cvt_transpose_bf16_kernel<<<(n + 255) / 256, 256, 0, stream>>>(W_in,  wWin,  DD,    NXZ);
        n = DIN * NDBL;    cvt_transpose_bf16_kernel<<<(n + 255) / 256, 256, 0, stream>>>(W_x,   wWx,   DIN,   NDBL);
        n = DTRANK * DIN;  cvt_transpose_bf16_kernel<<<(n + 255) / 256, 256, 0, stream>>>(W_dt,  wWdt,  DTRANK, DIN);
        n = DIN * DD;      cvt_transpose_bf16_kernel<<<(n + 255) / 256, 256, 0, stream>>>(W_out, wWout, DIN,   DD);
        n = DIN * DSTATE;  neg_exp_kernel<<<(n + 255) / 256, 256, 0, stream>>>(A_log, wA, n);
    }
    // 2) LN1 -> bf16
    ln1_kernel<<<MM / 8, 256, 0, stream>>>(x, ln1_w, ln1_b, xln);
    // 3) xz = x_ln @ W_in  (8192x512 * 512x2048)
    gemm_bf16_wmma<128, 128><<<dim3(NXZ / 128, MM / 128), 256, 0, stream>>>(
        xln, wWin, xzbuf, MM, NXZ, DD);
    // 4) causal depthwise conv + SiLU
    conv_silu_kernel<<<(MM * DIN + 255) / 256, 256, 0, stream>>>(
        xzbuf, conv_w, conv_b, xpbuf, xpb);
    // 5) dbl = xp @ W_x  (8192x1024 * 1024x64)
    gemm_bf16_wmma<128, 64><<<dim3(NDBL / 64, MM / 128), 256, 0, stream>>>(
        xpb, wWx, dblb, MM, NDBL, DIN);
    // 6) dt_r -> bf16
    extract_dtr_kernel<<<(MM * DTRANK + 255) / 256, 256, 0, stream>>>(dblb, dtr);
    // 7) dt_pre = dt_r @ W_dt  (8192x32 * 32x1024)
    gemm_bf16_wmma<128, 128><<<dim3(DIN / 128, MM / 128), 256, 0, stream>>>(
        dtr, wWdt, dtp, MM, DIN, DTRANK);
    // 8) selective scan (fused softplus, D-skip, z-gating) -> bf16
    scan_kernel<<<dim3(DIN / 256, BB), 256, 0, stream>>>(
        dtp, xpbuf, dblb, xzbuf, wA, D_skip, b_dt, ybuf);
    // 9) h = y @ W_out  (8192x1024 * 1024x512)
    gemm_bf16_wmma<128, 128><<<dim3(DD / 128, MM / 128), 256, 0, stream>>>(
        ybuf, wWout, hbuf, MM, DD, DIN);
    // 10) out = LN2(x + h)
    ln2_kernel<<<MM / 8, 256, 0, stream>>>(x, hbuf, ln2_w, ln2_b, out);
}
